// PDN_dense_add_skip_2bro_pre_batch_act_max_test_38010460569942
// MI455X (gfx1250) — matrix operand order repaired and checked by reading, not compiled, not run on hardware
//
#include <hip/hip_runtime.h>
#include <hip/hip_bf16.h>

// ---------------------------------------------------------------------------
// PDNConv GNN stack for MI455X (gfx1250).
// Memory/atomic-bound scatter dominates; node-transform GEMMs use
// v_wmma_f32_16x16x32_bf16. Hot tables (<70MB) stay L2-resident (192MB L2).
// ---------------------------------------------------------------------------

#define NNODES 50000
#define NEDGES 800000
#define FEAT   64
#define FEDGE  16
#define NGRAPH 512
#define NCLS   10
#define NLAYER 3
#define BN_EPS 1e-5f

typedef __attribute__((ext_vector_type(16))) __bf16 v16bf;
typedef __attribute__((ext_vector_type(8)))  float  v8f;

// ---------------- edge MLP: w = sigmoid(relu(ea@W1^T+b1)@W2^T+b2) -----------
__global__ void edge_mlp_kernel(const float* __restrict__ EA,
                                const float* __restrict__ W1,
                                const float* __restrict__ B1,
                                const float* __restrict__ W2,
                                const float* __restrict__ B2,
                                float* __restrict__ Wout) {
  int e = blockIdx.x * blockDim.x + threadIdx.x;
  if (e >= NEDGES) return;
  float ea[FEDGE];
  const float* p = EA + (size_t)e * FEDGE;
#pragma unroll
  for (int k = 0; k < FEDGE; ++k) ea[k] = p[k];
  float s = B2[0];
#pragma unroll
  for (int j = 0; j < FEDGE; ++j) {
    float h = B1[j];
#pragma unroll
    for (int k = 0; k < FEDGE; ++k) h = fmaf(ea[k], W1[j * FEDGE + k], h);
    h = fmaxf(h, 0.0f);
    s = fmaf(h, W2[j], s);
  }
  Wout[e] = 1.0f / (1.0f + __expf(-s));
}

// ---------------- degree / symmetric norm ----------------------------------
__global__ void deg_init_kernel(float* __restrict__ deg) {
  int i = blockIdx.x * blockDim.x + threadIdx.x;
  if (i < NNODES) deg[i] = 1.0f;  // self-loop weight
}

__global__ void deg_scatter_kernel(const int* __restrict__ col,
                                   const float* __restrict__ w,
                                   float* __restrict__ deg) {
  int e = blockIdx.x * blockDim.x + threadIdx.x;
  if (e < NEDGES) atomicAdd(&deg[col[e]], w[e]);
}

__global__ void deg_to_dis_kernel(float* __restrict__ deg) {
  int i = blockIdx.x * blockDim.x + threadIdx.x;
  if (i < NNODES) {
    float d = deg[i];
    deg[i] = (d > 0.0f) ? rsqrtf(d) : 0.0f;
  }
}

// coef[e] = dis[row]*w[e]*dis[col]  (in-place over w buffer)
__global__ void coef_kernel(const float* __restrict__ dis,
                            const int* __restrict__ row,
                            const int* __restrict__ col,
                            float* __restrict__ wcoef) {
  int e = blockIdx.x * blockDim.x + threadIdx.x;
  if (e < NEDGES) wcoef[e] = dis[row[e]] * wcoef[e] * dis[col[e]];
}

// ---------------- node transform GEMM: XL = X @ LW^T (WMMA bf16) -----------
// Fused epilogue also seeds the aggregation output with the self-loop term:
//   XA[n] = XL[n] * dis[n]^2    (self-loop norm = dis*1*dis)
// One wave computes one 16x16 output tile; 4 waves per block cover 64 cols.
// N = 50000 = 3125 * 16 -> no boundary masking, EXEC stays all-ones.
__global__ void __launch_bounds__(128)
gemm_node_kernel(const float* __restrict__ X,
                 const float* __restrict__ LW,
                 const float* __restrict__ dis,
                 float* __restrict__ XL,
                 float* __restrict__ XA, int nrows) {
  int wave = threadIdx.x >> 5;
  int lane = threadIdx.x & 31;
  int row0 = blockIdx.x * 16;
  int col0 = wave * 16;
  if (row0 >= nrows) return;               // uniform per wave

  int m = lane & 15;
  int khalf = lane >> 4;                   // 0: lanes 0-15, 1: lanes 16-31
  v8f acc = {};

#pragma unroll
  for (int k0 = 0; k0 < FEAT; k0 += 32) {
    v16bf a, b;
    // A fragment: 16-bit A 16x32 layout (ISA 7.12.2)
    const float* ap = X + (size_t)(row0 + m) * FEAT + k0 + khalf * 8;
#pragma unroll
    for (int i = 0; i < 8; ++i) {
      a[i]     = (__bf16)ap[i];            // K = kbase .. kbase+7
      a[i + 8] = (__bf16)ap[i + 16];       // K = kbase+16 .. kbase+23
    }
    // B fragment: B[k][n] = LW[col0+n][k]; same K order as A fragment
    const float* bp = LW + (size_t)(col0 + m) * FEAT + k0 + khalf * 8;
#pragma unroll
    for (int i = 0; i < 8; ++i) {
      b[i]     = (__bf16)bp[i];
      b[i + 8] = (__bf16)bp[i + 16];
    }

    acc = __builtin_amdgcn_wmma_f32_16x16x32_bf16(
        false, a, false, b, (short)0, acc, false, false);
  }

  // C/D layout: VGPR i -> (M = khalf*8 + i, N = lane&15)
  int n = lane & 15;
  int mb = row0 + khalf * 8;
#pragma unroll
  for (int i = 0; i < 8; ++i) {
    size_t off = (size_t)(mb + i) * FEAT + col0 + n;
    float v = acc[i];
    float d = dis[mb + i];                 // broadcast within half-wave
    XL[off] = v;
    XA[off] = v * d * d;                   // self-loop seed
  }
}

// ---------------- edge aggregation -----------------------------------------
// out[col] += xl[row] * coef[e]; 16 lanes per edge, float4 per lane
__global__ void conv_scatter_kernel(const float* __restrict__ XL,
                                    const float* __restrict__ coef,
                                    const int* __restrict__ row,
                                    const int* __restrict__ col,
                                    float* __restrict__ XO) {
  int t = blockIdx.x * blockDim.x + threadIdx.x;    // E*16 exact
  int e = t >> 4;
  int f = (t & 15) * 4;
  int r = row[e], c = col[e];
  float cf = coef[e];
  float4 v = *(const float4*)(XL + (size_t)r * FEAT + f);
  float* o = XO + (size_t)c * FEAT + f;
  atomicAdd(o + 0, v.x * cf);
  atomicAdd(o + 1, v.y * cf);
  atomicAdd(o + 2, v.z * cf);
  atomicAdd(o + 3, v.w * cf);
}

// ---------------- batch norm (training-mode stats) + relu ------------------
__global__ void stats_zero_kernel(float* __restrict__ stats) {
  if (threadIdx.x < 2 * FEAT) stats[threadIdx.x] = 0.0f;
}

__global__ void bn_stats_kernel(const float* __restrict__ X,
                                float* __restrict__ stats) {
  __shared__ float ssum[FEAT], ssq[FEAT];
  if (threadIdx.x < FEAT) { ssum[threadIdx.x] = 0.f; ssq[threadIdx.x] = 0.f; }
  __syncthreads();
  int ch = threadIdx.x & 63;
  int rl = threadIdx.x >> 6;                        // 4 rows per block pass
  float s = 0.f, q = 0.f;
  for (int rb = blockIdx.x * 4; rb < NNODES; rb += gridDim.x * 4) {
    int r = rb + rl;
    if (r < NNODES) {
      float v = X[(size_t)r * FEAT + ch];
      s += v; q += v * v;
    }
  }
  atomicAdd(&ssum[ch], s);
  atomicAdd(&ssq[ch], q);
  __syncthreads();
  if (threadIdx.x < FEAT) {
    atomicAdd(&stats[threadIdx.x], ssum[threadIdx.x]);
    atomicAdd(&stats[FEAT + threadIdx.x], ssq[threadIdx.x]);
  }
}

__global__ void bn_apply_relu_kernel(const float* __restrict__ X,
                                     const float* __restrict__ stats,
                                     float* __restrict__ Y) {
  int idx = blockIdx.x * blockDim.x + threadIdx.x;  // N*64 exact
  int ch = idx & 63;
  const float invN = 1.0f / (float)NNODES;
  float mu = stats[ch] * invN;
  float var = stats[FEAT + ch] * invN - mu * mu;    // biased var (matches ref)
  float y = (X[idx] - mu) * rsqrtf(var + BN_EPS);
  Y[idx] = fmaxf(y, 0.0f);
}

// ---------------- residual adds --------------------------------------------
__global__ void addskip_kernel(float* __restrict__ XA,
                               const float* __restrict__ X0,
                               float* __restrict__ X1) {
  int idx = blockIdx.x * blockDim.x + threadIdx.x;
  float t = XA[idx] + X0[idx];
  X1[idx] = t;
  XA[idx] = t;
}

// ---------------- global max pool (relu(x+x0+x1), segment max) -------------
__global__ void pool_init_kernel(unsigned* __restrict__ pooled) {
  int i = blockIdx.x * blockDim.x + threadIdx.x;
  if (i < NGRAPH * FEAT) pooled[i] = 0u;            // relu output >= 0
}

__global__ void pool_max_kernel(const float* __restrict__ XA,
                                const float* __restrict__ X0,
                                const float* __restrict__ X1,
                                const int* __restrict__ batch,
                                unsigned* __restrict__ pooled) {
  int idx = blockIdx.x * blockDim.x + threadIdx.x;  // N*64 exact
  int node = idx >> 6;
  int ch = idx & 63;
  float v = fmaxf(XA[idx] + X0[idx] + X1[idx], 0.0f);
  // non-negative IEEE floats: uint compare == float compare
  atomicMax(&pooled[(size_t)batch[node] * FEAT + ch], __float_as_uint(v));
}

// ---------------- final classifier: out = pooled @ lin_w^T + lin_b ---------
__global__ void classifier_kernel(const unsigned* __restrict__ pooled,
                                  const float* __restrict__ LWc,
                                  const float* __restrict__ LB,
                                  float* __restrict__ out) {
  int g = blockIdx.x;
  int k = threadIdx.x;
  if (k >= NCLS) return;
  float s = LB[k];
#pragma unroll
  for (int c = 0; c < FEAT; ++c)
    s = fmaf(__uint_as_float(pooled[(size_t)g * FEAT + c]), LWc[k * FEAT + c], s);
  out[(size_t)g * NCLS + k] = s;
}

// ---------------------------------------------------------------------------
extern "C" void kernel_launch(void* const* d_in, const int* in_sizes, int n_in,
                              void* d_out, int out_size, void* d_ws, size_t ws_size,
                              hipStream_t stream) {
  const float* x         = (const float*)d_in[0];
  const int*   eidx      = (const int*)d_in[1];
  const int*   batch     = (const int*)d_in[2];
  const float* edge_attr = (const float*)d_in[4];
  const float* lin_w     = (const float*)d_in[20];
  const float* lin_b     = (const float*)d_in[21];
  const int* row = eidx;
  const int* col = eidx + NEDGES;

  // workspace carve (floats)
  const size_t NC = (size_t)NNODES * FEAT;
  float* ws   = (float*)d_ws;
  float* XL   = ws;             // N*64 gemm output / gather table
  float* XA   = XL + NC;        // current node state
  float* XB   = XA + NC;        // bn+relu temp
  float* X0   = XB + NC;        // skip 0
  float* X1   = X0 + NC;        // skip 1
  float* Wbuf = X1 + NC;        // E edge weights -> edge coefs (per conv)
  float* Dis  = Wbuf + NEDGES;  // N  (deg -> rsqrt(deg))
  float* Stats = Dis + NNODES;  // 128
  unsigned* Pooled = (unsigned*)(Stats + 2 * FEAT);  // G*64

  const int TB = 256;
  const int gN    = (NNODES + TB - 1) / TB;
  const int gE    = (NEDGES + TB - 1) / TB;
  const int gNC   = (int)(NC / TB);             // 12500 exact
  const int gE16  = NEDGES * 16 / TB;           // 50000 exact
  const int gGemm = NNODES / 16;                // 3125 exact

  // per-conv parameter pointers: conv 0 = c1, 1..3 = h1[i], 4..6 = h2[i]
  const float *lwP[7], *w1P[7], *b1P[7], *w2P[7], *b2P[7];
  lwP[0] = (const float*)d_in[5];  w1P[0] = (const float*)d_in[6];
  b1P[0] = (const float*)d_in[7];  w2P[0] = (const float*)d_in[8];
  b2P[0] = (const float*)d_in[9];
  for (int i = 0; i < NLAYER; ++i) {
    lwP[1 + i] = (const float*)d_in[10] + (size_t)i * FEAT * FEAT;
    w1P[1 + i] = (const float*)d_in[11] + (size_t)i * FEDGE * FEDGE;
    b1P[1 + i] = (const float*)d_in[12] + (size_t)i * FEDGE;
    w2P[1 + i] = (const float*)d_in[13] + (size_t)i * FEDGE;
    b2P[1 + i] = (const float*)d_in[14] + (size_t)i;
    lwP[4 + i] = (const float*)d_in[15] + (size_t)i * FEAT * FEAT;
    w1P[4 + i] = (const float*)d_in[16] + (size_t)i * FEDGE * FEDGE;
    b1P[4 + i] = (const float*)d_in[17] + (size_t)i * FEDGE;
    w2P[4 + i] = (const float*)d_in[18] + (size_t)i * FEDGE;
    b2P[4 + i] = (const float*)d_in[19] + (size_t)i;
  }

  // runs one PDNConv: xin --(lw gemm + norm scatter)--> XA
  auto run_conv = [&](int ci, const float* xin) {
    edge_mlp_kernel<<<gE, TB, 0, stream>>>(edge_attr, w1P[ci], b1P[ci],
                                           w2P[ci], b2P[ci], Wbuf);
    deg_init_kernel<<<gN, TB, 0, stream>>>(Dis);
    deg_scatter_kernel<<<gE, TB, 0, stream>>>(col, Wbuf, Dis);
    deg_to_dis_kernel<<<gN, TB, 0, stream>>>(Dis);
    coef_kernel<<<gE, TB, 0, stream>>>(Dis, row, col, Wbuf);
    gemm_node_kernel<<<gGemm, 128, 0, stream>>>(xin, lwP[ci], Dis, XL, XA,
                                                NNODES);
    conv_scatter_kernel<<<gE16, TB, 0, stream>>>(XL, Wbuf, row, col, XA);
  };

  auto run_bn = [&](void) {  // XA -> XB = relu(batchnorm(XA))
    stats_zero_kernel<<<1, 128, 0, stream>>>(Stats);
    bn_stats_kernel<<<512, TB, 0, stream>>>(XA, Stats);
    bn_apply_relu_kernel<<<gNC, TB, 0, stream>>>(XA, Stats, XB);
  };

  // conv1
  run_conv(0, x);
  hipMemcpyAsync(X0, XA, NC * sizeof(float), hipMemcpyDeviceToDevice, stream);

  // hidden block 1
  for (int i = 0; i < NLAYER; ++i) { run_bn(); run_conv(1 + i, XB); }
  addskip_kernel<<<gNC, TB, 0, stream>>>(XA, X0, X1);   // x1 = x+x0; x = x+x0

  // hidden block 2
  for (int i = 0; i < NLAYER; ++i) { run_bn(); run_conv(4 + i, XB); }

  // pool + classify
  pool_init_kernel<<<(NGRAPH * FEAT + TB - 1) / TB, TB, 0, stream>>>(Pooled);
  pool_max_kernel<<<gNC, TB, 0, stream>>>(XA, X0, X1, batch, Pooled);
  classifier_kernel<<<NGRAPH, 32, 0, stream>>>(Pooled, lin_w, lin_b,
                                               (float*)d_out);
}
